// FDKFNet_82119774699578
// MI455X (gfx1250) — matrix operand order repaired and checked by reading, not compiled, not run on hardware
//
#include <hip/hip_runtime.h>
#include <hip/hip_bf16.h>

// ---------------------------------------------------------------------------
// FDKF-Net scan for MI455X (gfx1250, wave32, WMMA + TDM).
// One persistent 1024-thread workgroup runs the whole T=2000 recurrence.
// All matmuls -> v_wmma_f32_16x16x32_bf16. Per-step audio tiles streamed
// HBM -> LDS with TENSOR_LOAD_TO_LDS (double buffered, 1 step ahead).
// ---------------------------------------------------------------------------

#define NFREQ 481
#define FPAD  512
#define NB    64
#define HID   128
#define BATCH 8
#define TLEN  2000
#define NELEM (BATCH * NFREQ)   // 3848

typedef __attribute__((ext_vector_type(16))) __bf16 v16bf;
typedef __attribute__((ext_vector_type(8)))  __bf16 v8bf;
typedef __attribute__((ext_vector_type(8)))  float  v8f;
typedef __attribute__((ext_vector_type(4)))  unsigned int v4u;
typedef __attribute__((ext_vector_type(8)))  int v8i;
typedef __attribute__((ext_vector_type(4)))  int v4i;

// ws layout (in 16-bit units)
#define WS_WIH0 0         // 384*192
#define WS_WHH0 73728     // 384*128
#define WS_WIH1 122880    // 384*128
#define WS_WHH1 172032    // 384*128
#define WS_PROJ 221184    // 128*128
#define WS_MELT 237568    // 64*512 (mel_fb transposed, freq-padded)
#define WS_TOTAL 270336

static __device__ __forceinline__ unsigned short f2bf(float f) {
  unsigned int u = __float_as_uint(f);
  unsigned int r = (u + 0x7FFFu + ((u >> 16) & 1u)) >> 16;  // RNE
  return (unsigned short)r;
}

static __device__ __forceinline__ float sigf(float x) {
  return 1.0f / (1.0f + __expf(-x));
}

// 16-bit A fragment (16 rows x 32 K) from row-major bf16, per ISA layout:
// lanes 0-15 -> M rows with K pairs 0..7,16..23 ; lanes 16-31 -> K 8..15,24..31
static __device__ __forceinline__ v16bf load_a_frag(
    const unsigned short* __restrict__ base, int ld, int m0, int k0, int lane) {
  int m  = m0 + (lane & 15);
  int kb = k0 + ((lane >> 4) << 3);
  const unsigned short* p = base + m * ld + kb;
  union { v16bf v; v8bf h[2]; } u;
  u.h[0] = *reinterpret_cast<const v8bf*>(p);        // K = kb .. kb+7
  u.h[1] = *reinterpret_cast<const v8bf*>(p + 16);   // K = kb+16 .. kb+23
  return u.v;
}

// 16-bit B fragment (32 K x 16 N) from LDS stored [K][16] bf16: n = lane%16,
// lanes 0-15 -> K 0..7,16..23 ; lanes 16-31 -> K 8..15,24..31
static __device__ __forceinline__ v16bf load_b_frag(
    const unsigned short* base, int k0, int lane) {
  int n = lane & 15, h = lane >> 4;
  v16bf r;
  for (int s = 0; s < 16; ++s) {
    int k = (s < 8) ? (8 * h + s) : (16 + 8 * h + (s - 8));
    r[s] = __builtin_bit_cast(__bf16, base[(k0 + k) * 16 + n]);
  }
  return r;
}

static __device__ __forceinline__ v8f wmma_bf16(v16bf a, v16bf b, v8f c) {
  return __builtin_amdgcn_wmma_f32_16x16x32_bf16(
      /*neg_a=*/false, a, /*neg_b=*/false, b,
      /*c_mod=*/(short)0, c, /*reuse_a=*/false, /*reuse_b=*/false);
}

// TDM: async-load one 2D tile (BATCH rows x NFREQ f32, row stride TLEN*NFREQ
// elements) from global into LDS. D# layout per CDNA5 ISA ch.8.
static __device__ __forceinline__ void tdm_load_tile(const float* gsrc,
                                                     void* ldst) {
  unsigned long long ga = (unsigned long long)(uintptr_t)gsrc;
  unsigned int ldsOff = (unsigned int)(uintptr_t)ldst;  // LDS = addr[31:0]
  v4u g0;
  g0[0] = 1u;                                    // count=1, user descriptor
  g0[1] = ldsOff;                                // lds_addr (bytes)
  g0[2] = (unsigned int)ga;                      // global_addr[31:0]
  g0[3] = (unsigned int)((ga >> 32) & 0x01FFFFFFu) | (2u << 30);  // type=2
  v8i g1;
  g1[0] = 0x00020000;                            // data_size = 4B
  g1[1] = (int)((unsigned)NFREQ << 16);          // tensor_dim0[15:0]
  g1[2] = (int)((unsigned)BATCH << 16);          // tensor_dim1[15:0]
  g1[3] = (int)((unsigned)NFREQ << 16);          // tile_dim0
  g1[4] = (int)BATCH;                            // tile_dim1
  g1[5] = (int)(TLEN * NFREQ);                   // tensor_dim0_stride (elems)
  g1[6] = 0;
  g1[7] = 0;
  v4i z4 = {0, 0, 0, 0};
#if defined(__clang_major__) && (__clang_major__ >= 23)
  v8i z8 = {0, 0, 0, 0, 0, 0, 0, 0};
  __builtin_amdgcn_tensor_load_to_lds(g0, g1, z4, z4, z8, 0);
#else
  __builtin_amdgcn_tensor_load_to_lds(g0, g1, z4, z4, 0);
#endif
}

// ---------------------------------------------------------------------------
// Prep: convert weights + transposed mel filterbank to bf16 in workspace.
// ---------------------------------------------------------------------------
__global__ void fdkf_prep_kernel(const float* __restrict__ w_ih0,
                                 const float* __restrict__ w_hh0,
                                 const float* __restrict__ w_ih1,
                                 const float* __restrict__ w_hh1,
                                 const float* __restrict__ proj_w,
                                 const float* __restrict__ mel_fb,
                                 unsigned short* __restrict__ ws) {
  int gid = blockIdx.x * blockDim.x + threadIdx.x;
  if (gid < 73728) { ws[WS_WIH0 + gid] = f2bf(w_ih0[gid]); return; }
  gid -= 73728;
  if (gid < 49152) { ws[WS_WHH0 + gid] = f2bf(w_hh0[gid]); return; }
  gid -= 49152;
  if (gid < 49152) { ws[WS_WIH1 + gid] = f2bf(w_ih1[gid]); return; }
  gid -= 49152;
  if (gid < 49152) { ws[WS_WHH1 + gid] = f2bf(w_hh1[gid]); return; }
  gid -= 49152;
  if (gid < 16384) { ws[WS_PROJ + gid] = f2bf(proj_w[gid]); return; }
  gid -= 16384;
  if (gid < 32768) {
    int j = gid >> 9, f = gid & 511;  // melT[64][512]
    ws[WS_MELT + gid] = (f < NFREQ) ? f2bf(mel_fb[f * NB + j])
                                    : (unsigned short)0;
  }
}

// ---------------------------------------------------------------------------
// Persistent scan kernel: 1 block x 1024 threads (32 wave32s, 1 WGP).
// ---------------------------------------------------------------------------
__global__ __launch_bounds__(1024) void fdkf_scan_kernel(
    const float* __restrict__ mic_re, const float* __restrict__ mic_im,
    const float* __restrict__ ref_re, const float* __restrict__ ref_im,
    const float* __restrict__ mel_fb,
    const float* __restrict__ b_ih0, const float* __restrict__ b_hh0,
    const float* __restrict__ b_ih1, const float* __restrict__ b_hh1,
    const float* __restrict__ proj_b,
    const unsigned short* __restrict__ ws,
    float* __restrict__ out) {
  extern __shared__ unsigned char smem[];
  // audio staging: 2 buffers x 4 arrays x (BATCH*NFREQ) f32, filled by TDM
  float* audA = (float*)smem;                            // 4*3848 f32
  float* audB = audA + 4 * NELEM;                        // 4*3848 f32
  float* innR = audB + 4 * NELEM;                        // 3848 f32
  float* innI = innR + NELEM;                            // 3848 f32
  unsigned short* powMic = (unsigned short*)(innI + NELEM);  // [512][16] bf16
  unsigned short* powRef = powMic + FPAD * 16;           // [512][16]
  unsigned short* powInn = powRef + FPAD * 16;           // [512][16]
  unsigned short* featB  = powInn + FPAD * 16;           // [192][16]
  unsigned short* h0B    = featB + 192 * 16;             // [128][16]
  unsigned short* h1B    = h0B + HID * 16;               // [128][16]
  float* h0f = (float*)(h1B + HID * 16);                 // [128][8]
  float* h1f = h0f + HID * BATCH;                        // [128][8]
  float* gx  = h1f + HID * BATCH;                        // [384][16]
  float* gh  = gx + 384 * 16;                            // [384][16]
  float* cov = gh + 384 * 16;                            // [128][16]

  const unsigned short* w_ih0b = ws + WS_WIH0;
  const unsigned short* w_hh0b = ws + WS_WHH0;
  const unsigned short* w_ih1b = ws + WS_WIH1;
  const unsigned short* w_hh1b = ws + WS_WHH1;
  const unsigned short* projwb = ws + WS_PROJ;
  const unsigned short* melT   = ws + WS_MELT;

  const int tid  = threadIdx.x;
  const int lane = tid & 31;
  const int wv   = tid >> 5;

  // --- one-time LDS init (padded rows/cols must be zero) ---
  for (int i = tid; i < FPAD * 16; i += 1024) {
    powMic[i] = 0; powRef[i] = 0; powInn[i] = 0;
  }
  for (int i = tid; i < HID * 16; i += 1024) { h0B[i] = 0; h1B[i] = 0; }
  for (int i = tid; i < HID * BATCH; i += 1024) { h0f[i] = 0.f; h1f[i] = 0.f; }

  // --- per-thread Kalman state: 4 (b,f) elements per thread ---
  float Hre[4], Him[4], Pst[4];
  int bIdx[4], fIdx[4], base[4];
  bool val[4];
  for (int i = 0; i < 4; ++i) {
    int e = tid + i * 1024;
    val[i] = (e < NELEM);
    int b = val[i] ? (e / NFREQ) : 0;
    int f = val[i] ? (e % NFREQ) : 0;
    bIdx[i] = b; fIdx[i] = f;
    base[i] = b * TLEN * NFREQ + f;
    Hre[i] = 0.f; Him[i] = 0.f; Pst[i] = 0.1f;
  }

  // --- prologue: async-stage audio tiles for t=0 into buffer A ---
  if (wv == 0) {
    tdm_load_tile(mic_re, audA + 0 * NELEM);
    tdm_load_tile(mic_im, audA + 1 * NELEM);
    tdm_load_tile(ref_re, audA + 2 * NELEM);
    tdm_load_tile(ref_im, audA + 3 * NELEM);
  }
  __syncthreads();

  for (int t = 0; t < TLEN; ++t) {
    const float* aud = (t & 1) ? audB : audA;

    // ---- S0: wait for this step's TDM tiles, prefetch next step ----
    if (wv == 0) __builtin_amdgcn_s_wait_tensorcnt(0);
    __syncthreads();
    if (wv == 0 && (t + 1) < TLEN) {
      float* nb_ = (t & 1) ? audA : audB;
      const int goff = (t + 1) * NFREQ;
      tdm_load_tile(mic_re + goff, nb_ + 0 * NELEM);
      tdm_load_tile(mic_im + goff, nb_ + 1 * NELEM);
      tdm_load_tile(ref_re + goff, nb_ + 2 * NELEM);
      tdm_load_tile(ref_im + goff, nb_ + 3 * NELEM);
    }

    // ---- S1: powers + innovation (elementwise, all threads) ----
    for (int i = 0; i < 4; ++i) {
      if (!val[i]) continue;
      int e = tid + i * 1024;
      float mr = aud[e];
      float mi = aud[NELEM + e];
      float rr = aud[2 * NELEM + e];
      float ri = aud[3 * NELEM + e];
      float mp = mr * mr + mi * mi;
      float rp = rr * rr + ri * ri;
      float vr = mr - (Hre[i] * rr - Him[i] * ri);
      float vi = mi - (Hre[i] * ri + Him[i] * rr);
      innR[e] = vr;
      innI[e] = vi;
      int off = fIdx[i] * 16 + bIdx[i];
      powMic[off] = f2bf(mp);
      powRef[off] = f2bf(rp);
      powInn[off] = f2bf(vr * vr + vi * vi);
    }
    __syncthreads();

    // ---- S2: mel feature GEMMs (melT 64x512 @ powT 512x16), 12 waves ----
    if (wv < 12) {
      int buf = wv >> 2, mt = wv & 3;
      const unsigned short* pB = (buf == 0) ? powMic : (buf == 1) ? powRef : powInn;
      v8f acc = {};
      for (int kt = 0; kt < 16; ++kt) {
        v16bf a = load_a_frag(melT, FPAD, mt * 16, kt * 32, lane);
        v16bf b = load_b_frag(pB, kt * 32, lane);
        acc = wmma_bf16(a, b, acc);
      }
      int n = lane & 15, hi = lane >> 4;
      for (int r = 0; r < 8; ++r) {
        int band = mt * 16 + r + 8 * hi;
        featB[(buf * 64 + band) * 16 + n] = f2bf(log1pf(acc[r]));
      }
    }
    __syncthreads();

    // ---- S3: GRU0 gate GEMMs: gx = W_ih0 @ featT, gh = W_hh0 @ h0T ----
    if (wv < 24) {
      int m0 = wv * 16;
      v8f ax = {}, ah = {};
      for (int kt = 0; kt < 6; ++kt) {
        v16bf a = load_a_frag(w_ih0b, 192, m0, kt * 32, lane);
        v16bf b = load_b_frag(featB, kt * 32, lane);
        ax = wmma_bf16(a, b, ax);
      }
      for (int kt = 0; kt < 4; ++kt) {
        v16bf a = load_a_frag(w_hh0b, HID, m0, kt * 32, lane);
        v16bf b = load_b_frag(h0B, kt * 32, lane);
        ah = wmma_bf16(a, b, ah);
      }
      int n = lane & 15, hi = lane >> 4;
      for (int r = 0; r < 8; ++r) {
        int m = m0 + r + 8 * hi;
        gx[m * 16 + n] = ax[r] + b_ih0[m];
        gh[m * 16 + n] = ah[r] + b_hh0[m];
      }
    }
    __syncthreads();

    // ---- S4: GRU0 gates -> h0 (all 1024 threads: j = tid/8, b = tid%8) ----
    {
      int j = tid >> 3, b = tid & 7;
      float rg = sigf(gx[j * 16 + b] + gh[j * 16 + b]);
      float zg = sigf(gx[(HID + j) * 16 + b] + gh[(HID + j) * 16 + b]);
      float ng = tanhf(gx[(2 * HID + j) * 16 + b] + rg * gh[(2 * HID + j) * 16 + b]);
      float h = (1.f - zg) * ng + zg * h0f[j * 8 + b];
      h0f[j * 8 + b] = h;
      h0B[j * 16 + b] = f2bf(h);
    }
    __syncthreads();

    // ---- S5: GRU1 gate GEMMs ----
    if (wv < 24) {
      int m0 = wv * 16;
      v8f ax = {}, ah = {};
      for (int kt = 0; kt < 4; ++kt) {
        v16bf a = load_a_frag(w_ih1b, HID, m0, kt * 32, lane);
        v16bf b = load_b_frag(h0B, kt * 32, lane);
        ax = wmma_bf16(a, b, ax);
        v16bf a2 = load_a_frag(w_hh1b, HID, m0, kt * 32, lane);
        v16bf b2 = load_b_frag(h1B, kt * 32, lane);
        ah = wmma_bf16(a2, b2, ah);
      }
      int n = lane & 15, hi = lane >> 4;
      for (int r = 0; r < 8; ++r) {
        int m = m0 + r + 8 * hi;
        gx[m * 16 + n] = ax[r] + b_ih1[m];
        gh[m * 16 + n] = ah[r] + b_hh1[m];
      }
    }
    __syncthreads();

    // ---- S6: GRU1 gates -> h1 ----
    {
      int j = tid >> 3, b = tid & 7;
      float rg = sigf(gx[j * 16 + b] + gh[j * 16 + b]);
      float zg = sigf(gx[(HID + j) * 16 + b] + gh[(HID + j) * 16 + b]);
      float ng = tanhf(gx[(2 * HID + j) * 16 + b] + rg * gh[(2 * HID + j) * 16 + b]);
      float h = (1.f - zg) * ng + zg * h1f[j * 8 + b];
      h1f[j * 8 + b] = h;
      h1B[j * 16 + b] = f2bf(h);
    }
    __syncthreads();

    // ---- S7: projection GEMM: cov = proj_w @ h1T + proj_b ----
    if (wv < 8) {
      int m0 = wv * 16;
      v8f acc = {};
      for (int kt = 0; kt < 4; ++kt) {
        v16bf a = load_a_frag(projwb, HID, m0, kt * 32, lane);
        v16bf b = load_b_frag(h1B, kt * 32, lane);
        acc = wmma_bf16(a, b, acc);
      }
      int n = lane & 15, hi = lane >> 4;
      for (int r = 0; r < 8; ++r) {
        int m = m0 + r + 8 * hi;
        cov[m * 16 + n] = acc[r] + proj_b[m];
      }
    }
    __syncthreads();

    // ---- S8: Kalman update per (b,f) element (audio/innov from LDS) ----
    for (int i = 0; i < 4; ++i) {
      if (!val[i]) continue;
      int e = tid + i * 1024;
      float mr = aud[e];
      float mi = aud[NELEM + e];
      float rr = aud[2 * NELEM + e];
      float ri = aud[3 * NELEM + e];
      float mp = mr * mr + mi * mi;
      float rp = rr * rr + ri * ri;
      float vr = innR[e];
      float vi = innI[e];
      const float* mrow = mel_fb + fIdx[i] * NB;
      float dq = 0.f, dr = 0.f;
      for (int j = 0; j < NB; ++j) {
        float m = mrow[j];
        dq += cov[j * 16 + bIdx[i]] * m;
        dr += cov[(NB + j) * 16 + bIdx[i]] * m;
      }
      float Q = __expf(dq) + 1e-8f;
      float R = __expf(dr) + 1e-8f;
      float Pp = Pst[i] + Q;
      float S = rp * Pp + R;
      float inv = 1.f / S;
      float Kre = Pp * rr * inv;
      float Kim = -Pp * ri * inv;
      float vad = (mp < 20.f * (rp + 1e-8f)) ? 1.f : 0.f;
      Hre[i] += vad * (Kre * vr - Kim * vi);
      Him[i] += vad * (Kre * vi + Kim * vr);
      float krefre = Kre * rr - Kim * ri;
      Pst[i] = fminf(fmaxf((1.f - vad * krefre) * Pp, 1e-8f), 100.f);
      float sre = mr - (Hre[i] * rr - Him[i] * ri);
      float sim = mi - (Hre[i] * ri + Him[i] * rr);
      float* o = out + (size_t)(base[i] + t * NFREQ) * 2;
      o[0] = sre; o[1] = sim;
    }
    __syncthreads();  // cov/powT/innov are rewritten next iteration
  }

  // ---- epilogue: final states (H_f complex, P_f, stacked hidden) ----
  const size_t offH = (size_t)BATCH * TLEN * NFREQ * 2;
  const size_t offP = offH + (size_t)BATCH * NFREQ * 2;
  const size_t offG = offP + (size_t)BATCH * NFREQ;
  for (int i = 0; i < 4; ++i) {
    if (!val[i]) continue;
    size_t bf = (size_t)bIdx[i] * NFREQ + fIdx[i];
    out[offH + bf * 2]     = Hre[i];
    out[offH + bf * 2 + 1] = Him[i];
    out[offP + bf]         = Pst[i];
  }
  {
    int j = tid >> 3, b = tid & 7;
    out[offG + (size_t)b * HID + j] = h0f[j * 8 + b];
    out[offG + (size_t)BATCH * HID + (size_t)b * HID + j] = h1f[j * 8 + b];
  }
}

// ---------------------------------------------------------------------------
extern "C" void kernel_launch(void* const* d_in, const int* in_sizes, int n_in,
                              void* d_out, int out_size, void* d_ws, size_t ws_size,
                              hipStream_t stream) {
  (void)in_sizes; (void)n_in; (void)out_size; (void)ws_size;
  const float* mic_re = (const float*)d_in[0];
  const float* mic_im = (const float*)d_in[1];
  const float* ref_re = (const float*)d_in[2];
  const float* ref_im = (const float*)d_in[3];
  const float* mel_fb = (const float*)d_in[4];
  const float* w_ih0  = (const float*)d_in[5];
  const float* w_hh0  = (const float*)d_in[6];
  const float* b_ih0  = (const float*)d_in[7];
  const float* b_hh0  = (const float*)d_in[8];
  const float* w_ih1  = (const float*)d_in[9];
  const float* w_hh1  = (const float*)d_in[10];
  const float* b_ih1  = (const float*)d_in[11];
  const float* b_hh1  = (const float*)d_in[12];
  const float* proj_w = (const float*)d_in[13];
  const float* proj_b = (const float*)d_in[14];
  unsigned short* ws = (unsigned short*)d_ws;

  fdkf_prep_kernel<<<(WS_TOTAL + 255) / 256, 256, 0, stream>>>(
      w_ih0, w_hh0, w_ih1, w_hh1, proj_w, mel_fb, ws);

  const size_t smem =
      2 * 4 * NELEM * sizeof(float) +              // audio double buffer (TDM)
      2 * NELEM * sizeof(float) +                  // innov re/im
      3 * (FPAD * 16 * sizeof(unsigned short)) +   // pow buffers
      192 * 16 * sizeof(unsigned short) +          // featB
      2 * (HID * 16 * sizeof(unsigned short)) +    // h0B, h1B
      2 * (HID * BATCH * sizeof(float)) +          // h0f, h1f
      2 * (384 * 16 * sizeof(float)) +             // gx, gh
      HID * 16 * sizeof(float);                    // cov  (= 282944 B)

  hipFuncSetAttribute(reinterpret_cast<const void*>(fdkf_scan_kernel),
                      hipFuncAttributeMaxDynamicSharedMemorySize, (int)smem);

  fdkf_scan_kernel<<<1, 1024, smem, stream>>>(
      mic_re, mic_im, ref_re, ref_im, mel_fb,
      b_ih0, b_hh0, b_ih1, b_hh1, proj_b, ws, (float*)d_out);
}
